// SelfAttention_70068096467041
// MI455X (gfx1250) — matrix operand order repaired and checked by reading, not compiled, not run on hardware
//
#include <hip/hip_runtime.h>

typedef __bf16 bf16;
typedef __attribute__((ext_vector_type(8)))  __bf16 v8bf;
typedef __attribute__((ext_vector_type(16))) __bf16 v16bf;
typedef __attribute__((ext_vector_type(8)))  float  v8f;
typedef __attribute__((ext_vector_type(4)))  int    v4i;

#define BB 4
#define NN 4096
#define DD 256
#define ROWS (BB*NN)
#define SCALE 0.0625f   // D^-0.5, folded into Q at QKV epilogue

union V16 { v16bf v; v8bf h[2]; };

__device__ __forceinline__ v8f wmma_bf16(v16bf a, v16bf b, v8f c) {
  return __builtin_amdgcn_wmma_f32_16x16x32_bf16(false, a, false, b, (short)0, c, false, false);
}

// 16-byte async copy global -> LDS (ASYNCcnt-tracked, no VGPR round trip)
__device__ __forceinline__ void async_cp16(bf16* lds, const bf16* g) {
#if __has_builtin(__builtin_amdgcn_global_load_async_to_lds_b128)
  bf16* gnc = const_cast<bf16*>(g);
  __builtin_amdgcn_global_load_async_to_lds_b128(
      (__attribute__((address_space(1))) v4i*)gnc,
      (__attribute__((address_space(3))) v4i*)lds, 0, 0);
#else
  unsigned loff = (unsigned)(size_t)(__attribute__((address_space(3))) void*)lds;
  asm volatile("global_load_async_to_lds_b128 %0, %1, off" :: "v"(loff), "v"(g) : "memory");
#endif
}

__device__ __forceinline__ void wait_async0() {
#if __has_builtin(__builtin_amdgcn_s_wait_asynccnt)
  __builtin_amdgcn_s_wait_asynccnt(0);
#else
  asm volatile("s_wait_asynccnt 0x0" ::: "memory");
#endif
}

__device__ __forceinline__ void sched_fence() {
#if __has_builtin(__builtin_amdgcn_sched_barrier)
  __builtin_amdgcn_sched_barrier(0);
#endif
}

// load a B fragment: column n = lane%16 (in p), k in table order, row stride ld
__device__ __forceinline__ v16bf load_bfrag(const bf16* p, int ld) {
  V16 u;
  int hk = (threadIdx.x & 16) ? 8 : 0;
#pragma unroll
  for (int s = 0; s < 8; s++) {
    u.v[s] = p[(size_t)(hk + s) * ld];
    u.v[8 + s] = p[(size_t)(16 + hk + s) * ld];
  }
  return u.v;
}

// ---------------- LayerNorm: one wave per row, 8 elems/lane ----------------
__global__ void ln_kernel(const float* __restrict__ x, const float* __restrict__ g,
                          const float* __restrict__ bta, bf16* __restrict__ nx) {
  int w = threadIdx.x >> 5, lane = threadIdx.x & 31;
  int row = blockIdx.x * (blockDim.x >> 5) + w;
  const float* xr = x + (size_t)row * DD + lane * 8;
  float v[8];
  float s = 0.f, sq = 0.f;
#pragma unroll
  for (int i = 0; i < 8; i++) { v[i] = xr[i]; s += v[i]; sq += v[i]*v[i]; }
#pragma unroll
  for (int m = 1; m < 32; m <<= 1) { s += __shfl_xor(s, m, 32); sq += __shfl_xor(sq, m, 32); }
  float mu = s * (1.0f / DD);
  float var = sq * (1.0f / DD) - mu * mu;
  float rs = rsqrtf(var + 1e-5f);
  v8bf o;
#pragma unroll
  for (int i = 0; i < 8; i++) {
    int d = lane * 8 + i;
    o[i] = (bf16)((v[i] - mu) * rs * g[d] + bta[d]);
  }
  *(v8bf*)(nx + (size_t)row * DD + lane * 8) = o;
}

// ---------------- fp32 -> bf16 convert ----------------
__global__ void cvt_kernel(const float* __restrict__ src, bf16* __restrict__ dst, int n) {
  int i = blockIdx.x * blockDim.x + threadIdx.x;
  if (i < n) dst[i] = (bf16)src[i];
}

// ---------------- QKV GEMM: nx[16384,256] x wq[256,768]; +coord on q,k ----------------
// Q (pre-scaled by D^-0.5), V row-major [16384][256]; K TRANSPOSED: Kt[b][d][n]
__global__ __launch_bounds__(128) void qkv_kernel(const bf16* __restrict__ nx,
                                                  const bf16* __restrict__ wq,
                                                  const float* __restrict__ coord,
                                                  bf16* __restrict__ Qb, bf16* __restrict__ KtG,
                                                  bf16* __restrict__ Vb) {
  int w = threadIdx.x >> 5, lane = threadIdx.x & 31;
  int m0 = blockIdx.x * 64 + w * 16;
  int mrow = m0 + (lane & 15);
  int off = (lane & 16) ? 8 : 0;
  int mcol = lane & 15;
  int rbase = (lane & 16) ? 8 : 0;
  V16 a[8];
#pragma unroll
  for (int kc = 0; kc < 8; kc++) {
    const bf16* p = nx + (size_t)mrow * DD + kc * 32 + off;
    a[kc].h[0] = *(const v8bf*)p;
    a[kc].h[1] = *(const v8bf*)(p + 16);
  }
  v8f zero = {};
  for (int t = 0; t < 48; t++) {
    int e0 = t * 16;
    // batch all 8 B-fragment loads, then run the 8 WMMAs back-to-back
    v16bf bfr[8];
#pragma unroll
    for (int kc = 0; kc < 8; kc++)
      bfr[kc] = load_bfrag(wq + (size_t)(kc * 32) * 768 + e0 + mcol, 768);
    sched_fence();
    v8f acc = zero;
#pragma unroll
    for (int kc = 0; kc < 8; kc++) acc = wmma_bf16(a[kc].v, bfr[kc], acc);
    sched_fence();

    int colin = (e0 & 255) + mcol;       // column within the 256-wide chunk
    if (t < 16) {                        // ---- Q: row-major, + coord, * SCALE
#pragma unroll
      for (int r = 0; r < 8; r++) {
        int row = m0 + rbase + r;
        Qb[(size_t)row * DD + colin] =
            (bf16)((acc[r] + coord[(size_t)row * DD + colin]) * SCALE);
      }
    } else if (t < 32) {                 // ---- K: transposed layout + coord, packed 16B store
      v8bf kv;
#pragma unroll
      for (int r = 0; r < 8; r++) {
        int row = m0 + rbase + r;
        kv[r] = (bf16)(acc[r] + coord[(size_t)row * DD + colin]);
      }
      int bidx = m0 >> 12;
      int nbase = (m0 & (NN - 1)) + rbase;
      *(v8bf*)(KtG + ((size_t)bidx * DD + colin) * NN + nbase) = kv;
    } else {                             // ---- V: row-major
#pragma unroll
      for (int r = 0; r < 8; r++) {
        int row = m0 + rbase + r;
        Vb[(size_t)row * DD + colin] = (bf16)acc[r];
      }
    }
  }
}

// ---------------- Flash attention: WG = 64 query rows, wave = 16 rows ----------------
#define KT_S 72    // padded stride for K-transposed tile [256][64]
#define V_S  264   // padded stride for V tile [64][256]
#define P_S  72    // padded stride for P scratch [16][64]

__global__ __launch_bounds__(128) void attn_kernel(const bf16* __restrict__ Qb,
                                                   const bf16* __restrict__ KtG,
                                                   const bf16* __restrict__ Vb,
                                                   bf16* __restrict__ Ob) {
  __shared__ bf16 kT[2 * 256 * KT_S];   // 73728 B
  __shared__ bf16 vS[2 * 64 * V_S];     // 67584 B
  __shared__ bf16 pS[4 * 16 * P_S];     //  9216 B

  int w = threadIdx.x >> 5, lane = threadIdx.x & 31, tid = threadIdx.x;
  int b = blockIdx.x >> 6;     // 64 q-tiles per batch
  int qt = blockIdx.x & 63;
  int rowg = b * NN + qt * 64 + w * 16 + (lane & 15);
  int off = (lane & 16) ? 8 : 0;
  int m = lane & 15;
  int rbase = (lane & 16) ? 8 : 0;

  // Q A-fragments resident in registers (already scaled by D^-0.5)
  V16 qa[8];
#pragma unroll
  for (int kc = 0; kc < 8; kc++) {
    const bf16* p = Qb + (size_t)rowg * DD + kc * 32 + off;
    qa[kc].h[0] = *(const v8bf*)p;
    qa[kc].h[1] = *(const v8bf*)(p + 16);
  }

  // async stage of one key tile (K transposed rows d, V row-major) into buffer `buf`
  auto issue_tile = [&](int kt, int buf) {
    const bf16* kg = KtG + (size_t)b * DD * NN + (size_t)kt * 64;
    const bf16* vg = Vb + ((size_t)b * NN + (size_t)kt * 64) * DD;
    bf16* kd = kT + buf * 256 * KT_S;
    bf16* vd = vS + buf * 64 * V_S;
    for (int idx = tid; idx < 2048; idx += 128) {
      int d = idx >> 3, n8 = (idx & 7) * 8;          // K: 256 rows x 64 cols
      async_cp16(kd + d * KT_S + n8, kg + (size_t)d * NN + n8);
      int vr = idx >> 5, vd8 = (idx & 31) * 8;       // V: 64 rows x 256 cols
      async_cp16(vd + vr * V_S + vd8, vg + vr * DD + vd8);
    }
  };

  float mrow[8], lrow[8];
  v8f o[16];
  v8f zero = {};
#pragma unroll
  for (int r = 0; r < 8; r++) { mrow[r] = -__builtin_inff(); lrow[r] = 0.f; }
#pragma unroll
  for (int c = 0; c < 16; c++) o[c] = zero;

  bf16* pW = pS + w * 16 * P_S;

  issue_tile(0, 0);
  for (int kt = 0; kt < 64; kt++) {
    int buf = kt & 1;
    wait_async0();
    __syncthreads();
    if (kt + 1 < 64) issue_tile(kt + 1, buf ^ 1);

    const bf16* kTb = kT + buf * 256 * KT_S;
    const bf16* vSb = vS + buf * 64 * V_S;

    // S = Q K^T (4 tiles of 16x16, accumulated over 8 k-chunks of 32)
    v8f s[4];
#pragma unroll
    for (int t = 0; t < 4; t++) s[t] = zero;
#pragma unroll
    for (int kc = 0; kc < 8; kc++) {
      v16bf bfr[4];
#pragma unroll
      for (int t = 0; t < 4; t++)
        bfr[t] = load_bfrag(kTb + (kc * 32) * KT_S + t * 16 + m, KT_S);
      sched_fence();
#pragma unroll
      for (int t = 0; t < 4; t++) s[t] = wmma_bf16(qa[kc].v, bfr[t], s[t]);
      sched_fence();
    }

    // online softmax per row (rows split across lane half-groups)
    float alpha[8];
#pragma unroll
    for (int r = 0; r < 8; r++) {
      float mx = -__builtin_inff();
#pragma unroll
      for (int t = 0; t < 4; t++) mx = fmaxf(mx, s[t][r]);
#pragma unroll
      for (int msk = 1; msk < 16; msk <<= 1) mx = fmaxf(mx, __shfl_xor(mx, msk, 32));
      float mnew = fmaxf(mrow[r], mx);
      alpha[r] = __expf(mrow[r] - mnew);
      mrow[r] = mnew;
      float psum = 0.f;
#pragma unroll
      for (int t = 0; t < 4; t++) {
        float p = __expf(s[t][r] - mnew);
        s[t][r] = p;
        psum += p;
      }
#pragma unroll
      for (int msk = 1; msk < 16; msk <<= 1) psum += __shfl_xor(psum, msk, 32);
      lrow[r] = lrow[r] * alpha[r] + psum;
    }
#pragma unroll
    for (int c = 0; c < 16; c++)
#pragma unroll
      for (int r = 0; r < 8; r++) o[c][r] *= alpha[r];

    // C-layout -> A-layout for P via per-wave LDS scratch
#pragma unroll
    for (int t = 0; t < 4; t++)
#pragma unroll
      for (int r = 0; r < 8; r++)
        pW[(r + rbase) * P_S + t * 16 + m] = (bf16)s[t][r];
    __builtin_amdgcn_fence(__ATOMIC_RELEASE, "wavefront");
    __builtin_amdgcn_wave_barrier();
    __builtin_amdgcn_fence(__ATOMIC_ACQUIRE, "wavefront");

    V16 pa[2];
#pragma unroll
    for (int jc = 0; jc < 2; jc++) {
      pa[jc].h[0] = *(const v8bf*)(pW + m * P_S + jc * 32 + off);
      pa[jc].h[1] = *(const v8bf*)(pW + m * P_S + jc * 32 + 16 + off);
    }
    // O += P V
#pragma unroll
    for (int c = 0; c < 16; c += 4) {
      v16bf bfr[8];
#pragma unroll
      for (int cc = 0; cc < 4; cc++)
#pragma unroll
        for (int jc = 0; jc < 2; jc++)
          bfr[cc * 2 + jc] = load_bfrag(vSb + (jc * 32) * V_S + (c + cc) * 16 + m, V_S);
      sched_fence();
#pragma unroll
      for (int cc = 0; cc < 4; cc++)
#pragma unroll
        for (int jc = 0; jc < 2; jc++)
          o[c + cc] = wmma_bf16(pa[jc].v, bfr[cc * 2 + jc], o[c + cc]);
      sched_fence();
    }
  }

  // epilogue: O /= l, store bf16
  int rowbase = b * NN + qt * 64 + w * 16;
#pragma unroll
  for (int r = 0; r < 8; r++) {
    float inv = 1.0f / lrow[r];
    int row = rowbase + rbase + r;
#pragma unroll
    for (int c = 0; c < 16; c++)
      Ob[(size_t)row * DD + c * 16 + m] = (bf16)(o[c][r] * inv);
  }
}

// ---------------- Output GEMM: attn[16384,256] x w_out[256,256] + b_out ----------------
__global__ __launch_bounds__(128) void out_kernel(const bf16* __restrict__ A,
                                                  const bf16* __restrict__ Wt,
                                                  const float* __restrict__ bias,
                                                  float* __restrict__ out) {
  int w = threadIdx.x >> 5, lane = threadIdx.x & 31;
  int m0 = blockIdx.x * 64 + w * 16;
  int mrow = m0 + (lane & 15);
  int off = (lane & 16) ? 8 : 0;
  V16 a[8];
#pragma unroll
  for (int kc = 0; kc < 8; kc++) {
    const bf16* p = A + (size_t)mrow * DD + kc * 32 + off;
    a[kc].h[0] = *(const v8bf*)p;
    a[kc].h[1] = *(const v8bf*)(p + 16);
  }
  v8f zero = {};
  for (int t = 0; t < 16; t++) {
    int e0 = t * 16;
    v16bf bfr[8];
#pragma unroll
    for (int kc = 0; kc < 8; kc++)
      bfr[kc] = load_bfrag(Wt + (size_t)(kc * 32) * DD + e0 + (lane & 15), DD);
    sched_fence();
    v8f acc = zero;
#pragma unroll
    for (int kc = 0; kc < 8; kc++) acc = wmma_bf16(a[kc].v, bfr[kc], acc);
    sched_fence();

    int col = e0 + (lane & 15);
    int rbase = (lane & 16) ? 8 : 0;
    float bv = bias[col];
#pragma unroll
    for (int r = 0; r < 8; r++)
      out[(size_t)(m0 + rbase + r) * DD + col] = acc[r] + bv;
  }
}

extern "C" void kernel_launch(void* const* d_in, const int* in_sizes, int n_in,
                              void* d_out, int out_size, void* d_ws, size_t ws_size,
                              hipStream_t stream) {
  const float* x     = (const float*)d_in[0];
  const float* coord = (const float*)d_in[1];
  const float* ln_g  = (const float*)d_in[2];
  const float* ln_b  = (const float*)d_in[3];
  const float* w_qkv = (const float*)d_in[4];
  const float* w_out = (const float*)d_in[5];
  const float* b_out = (const float*)d_in[6];
  float* out = (float*)d_out;

  char* ws = (char*)d_ws;
  const size_t mat = (size_t)ROWS * DD * sizeof(bf16);   // 8 MB each
  bf16* nx  = (bf16*)ws; ws += mat;
  bf16* Qb  = (bf16*)ws; ws += mat;
  bf16* Ktg = (bf16*)ws; ws += mat;   // K transposed: [B][256][4096]
  bf16* Vb  = (bf16*)ws; ws += mat;
  bf16* Ab  = (bf16*)ws; ws += mat;
  bf16* wqb = (bf16*)ws; ws += (size_t)256 * 768 * sizeof(bf16);
  bf16* wob = (bf16*)ws; ws += (size_t)256 * 256 * sizeof(bf16);

  ln_kernel<<<ROWS / 8, 256, 0, stream>>>(x, ln_g, ln_b, nx);
  cvt_kernel<<<(256 * 768 + 255) / 256, 256, 0, stream>>>(w_qkv, wqb, 256 * 768);
  cvt_kernel<<<(256 * 256 + 255) / 256, 256, 0, stream>>>(w_out, wob, 256 * 256);
  qkv_kernel<<<ROWS / 64, 128, 0, stream>>>(nx, wqb, coord, Qb, Ktg, Vb);
  attn_kernel<<<BB * (NN / 64), 128, 0, stream>>>(Qb, Ktg, Vb, Ab);
  out_kernel<<<ROWS / 64, 128, 0, stream>>>(Ab, wob, b_out, out);
}
